// FullAttention_541165879331
// MI455X (gfx1250) — matrix operand order repaired and checked by reading, not compiled, not run on hardware
//
#include <hip/hip_runtime.h>

// ---------- problem constants ----------
#define PB 4       // batch
#define PL 2048    // sequence length
#define PH 16      // heads
#define PE 64      // head dim (E == D)
// scale * log2(e): softmax done in exp2 domain, folded into Q at bf16-convert
#define SC2 0.1803368801111243f

typedef __bf16 v16bf __attribute__((ext_vector_type(16)));
typedef __bf16 v8bf  __attribute__((ext_vector_type(8)));
typedef float  v8f   __attribute__((ext_vector_type(8)));

union Pack16 { v16bf v; v8bf h8[2]; __bf16 e[16]; };

#if __has_builtin(__builtin_amdgcn_exp2f)
#define FAST_EXP2(x) __builtin_amdgcn_exp2f(x)   // bare v_exp_f32; args <= 0
#else
#define FAST_EXP2(x) exp2f(x)
#endif
#if __has_builtin(__builtin_amdgcn_rcpf)
#define FAST_RCP(x) __builtin_amdgcn_rcpf(x)
#else
#define FAST_RCP(x) (1.0f / (x))
#endif

// xor-swizzle max within 16-lane group (and_mask=0x1f, or=0, xor=IMM>>10)
template <int IMM>
static __device__ __forceinline__ float swzmax(float x) {
    int y = __builtin_amdgcn_ds_swizzle(__builtin_bit_cast(int, x), IMM);
    return fmaxf(x, __builtin_bit_cast(float, y));
}
#define SWZ_XOR(m) (((m) << 10) | 0x1f)

// pack two f32 -> bf16 pair in one dword
static __device__ __forceinline__ unsigned pkbf(float a, float b) {
    unsigned x = (unsigned)__builtin_bit_cast(unsigned short, (__bf16)a);
    unsigned y = (unsigned)__builtin_bit_cast(unsigned short, (__bf16)b);
    return x | (y << 16);
}

// One workgroup = 8 waves * 32 lanes; wave w owns query rows [q0, q0+16).
// S streamed in 64-column blocks, double-buffered in LDS.
extern "C" __global__ void __launch_bounds__(256)
fa_causal_wmma_bf16(const float* __restrict__ Qg,
                    const float* __restrict__ Kg,
                    const float* __restrict__ Vg,
                    float* __restrict__ Out) {
    __shared__ __attribute__((aligned(16))) __bf16 Kl[2][64][72];   // [buf][s][e]
    __shared__ __attribute__((aligned(16))) __bf16 Vt[2][64][72];   // [buf][d][s]
    __shared__ __attribute__((aligned(16))) __bf16 Pl[8][16][72];   // [wave][m][k]

    const int tid  = threadIdx.x;
    const int wave = tid >> 5;
    const int lane = tid & 31;
    const int lo   = lane & 15;
    const int hi   = lane >> 4;

    const int nQB  = PL / 128;
    const int qb   = blockIdx.x % nQB;
    const int bh   = blockIdx.x / nQB;
    const int b    = bh / PH;
    const int h    = bh % PH;
    const int q0wg = qb * 128;
    const int q0   = q0wg + wave * 16;

    // ---- Q fragment, pre-scaled by scale*log2e, A-layout 16x64 bf16 ----
    Pack16 qa[2];
#pragma unroll
    for (int c = 0; c < 2; ++c)
#pragma unroll
        for (int i = 0; i < 16; ++i) {
            const int e = c * 32 + (hi << 4) + i;
            const size_t off = ((size_t)((size_t)b * PL + (q0 + lo)) * PH + h) * PE + e;
            qa[c].e[i] = (__bf16)(Qg[off] * SC2);
        }

    // ones B-fragment for rowsum-by-WMMA
    Pack16 ones;
#pragma unroll
    for (int i = 0; i < 16; ++i)
        ones.e[i] = __builtin_bit_cast(__bf16, (unsigned short)0x3F80);

    v8f o[4];
#pragma unroll
    for (int t = 0; t < 4; ++t) o[t] = (v8f){0, 0, 0, 0, 0, 0, 0, 0};
    float mrow[8], lrow[8];
#pragma unroll
    for (int r = 0; r < 8; ++r) { mrow[r] = -1e30f; lrow[r] = 0.0f; }

    // tile loader: 64x64 f32 -> bf16; K row-major (packed b32 stores),
    // V transposed with 2d x 8s per thread so stores pack along s.
    auto loadTile = [&](int buf, int s0) {
        {   // K: thread owns 1 s-row x 16 e
            const int s  = tid >> 2;
            const int e0 = (tid & 3) * 16;
            const float* kp = Kg + ((size_t)((size_t)b * PL + (s0 + s)) * PH + h) * PE + e0;
#pragma unroll
            for (int j = 0; j < 8; ++j)
                *(unsigned*)&Kl[buf][s][e0 + 2 * j] = pkbf(kp[2 * j], kp[2 * j + 1]);
        }
        {   // V: thread owns 2 d-cols x 8 s-rows
            const int s8 = wave * 8;            // 8 rows per wave
            const int d0 = lane * 2;            // 2 cols per lane
            float vv[8][2];
#pragma unroll
            for (int j = 0; j < 8; ++j) {
                const float* vp = Vg + ((size_t)((size_t)b * PL + (s0 + s8 + j)) * PH + h) * PE + d0;
                vv[j][0] = vp[0];
                vv[j][1] = vp[1];
            }
#pragma unroll
            for (int u = 0; u < 4; ++u) {
                *(unsigned*)&Vt[buf][d0][s8 + 2 * u]     = pkbf(vv[2 * u][0], vv[2 * u + 1][0]);
                *(unsigned*)&Vt[buf][d0 + 1][s8 + 2 * u] = pkbf(vv[2 * u][1], vv[2 * u + 1][1]);
            }
        }
    };

    const int nblk = (q0wg + 128) >> 6;    // causal upper bound for this WG
    loadTile(0, 0);
    __syncthreads();

    for (int blk = 0; blk < nblk; ++blk) {
        const int cur = blk & 1;
        if (blk + 1 < nblk) loadTile(cur ^ 1, (blk + 1) << 6);

        const int s0 = blk << 6;
        if (s0 <= q0 + 15) {               // uniform per wave
            // ---- S = Q*K^T : 4 tiles of 16x16, K-dim 64 ----
            v8f st[4];
#pragma unroll
            for (int t = 0; t < 4; ++t) {
                st[t] = (v8f){0, 0, 0, 0, 0, 0, 0, 0};
#pragma unroll
                for (int c = 0; c < 2; ++c) {
                    Pack16 kb;
                    const v8bf* p = (const v8bf*)&Kl[cur][t * 16 + lo][c * 32 + hi * 16];
                    kb.h8[0] = p[0]; kb.h8[1] = p[1];
                    st[t] = __builtin_amdgcn_wmma_f32_16x16x32_bf16(
                        false, qa[c].v, false, kb.v, (short)0, st[t], false, false);
                }
            }

            // ---- causal mask: only the diagonal-overlapping block ----
            if (s0 + 64 > q0) {
#pragma unroll
                for (int t = 0; t < 4; ++t) {
                    const int col = s0 + t * 16 + lo;
#pragma unroll
                    for (int r = 0; r < 8; ++r)
                        if (col > q0 + r + hi * 8) st[t][r] = -1e30f;
                }
            }

            // ---- row max: 3 VALU max + 4 xor-swizzle steps per row ----
            float rm[8], alpha[8];
#pragma unroll
            for (int r = 0; r < 8; ++r)
                rm[r] = fmaxf(fmaxf(st[0][r], st[1][r]), fmaxf(st[2][r], st[3][r]));
#pragma unroll
            for (int r = 0; r < 8; ++r) {
                rm[r] = swzmax<SWZ_XOR(1)>(rm[r]);
                rm[r] = swzmax<SWZ_XOR(2)>(rm[r]);
                rm[r] = swzmax<SWZ_XOR(4)>(rm[r]);
                rm[r] = swzmax<SWZ_XOR(8)>(rm[r]);
            }
#pragma unroll
            for (int r = 0; r < 8; ++r) {
                const float mnew = fmaxf(mrow[r], rm[r]);
                alpha[r] = FAST_EXP2(mrow[r] - mnew);
                mrow[r]  = mnew;
                lrow[r] *= alpha[r];
            }
#pragma unroll
            for (int t = 0; t < 4; ++t)
#pragma unroll
                for (int r = 0; r < 8; ++r) st[t][r] = FAST_EXP2(st[t][r] - mrow[r]);
#pragma unroll
            for (int t = 0; t < 4; ++t)
#pragma unroll
                for (int r = 0; r < 8; ++r) o[t][r] *= alpha[r];

            // ---- stage P (C-layout -> A-layout) in wave-private LDS ----
#pragma unroll
            for (int t = 0; t < 4; ++t)
#pragma unroll
                for (int r = 0; r < 8; ++r)
                    Pl[wave][r + 8 * hi][t * 16 + lo] = (__bf16)st[t][r];
            asm volatile("s_wait_dscnt 0" ::: "memory");

            Pack16 pa[2];
#pragma unroll
            for (int c = 0; c < 2; ++c) {
                pa[c].h8[0] = *(const v8bf*)&Pl[wave][lo][c * 32 + 16 * hi];
                pa[c].h8[1] = *(const v8bf*)&Pl[wave][lo][c * 32 + 16 * hi + 8];
            }

            // ---- row sum via WMMA with ones: replicated across C-layout ----
            v8f rsv = (v8f){0, 0, 0, 0, 0, 0, 0, 0};
            rsv = __builtin_amdgcn_wmma_f32_16x16x32_bf16(
                false, pa[0].v, false, ones.v, (short)0, rsv, false, false);
            rsv = __builtin_amdgcn_wmma_f32_16x16x32_bf16(
                false, pa[1].v, false, ones.v, (short)0, rsv, false, false);
#pragma unroll
            for (int r = 0; r < 8; ++r) lrow[r] += rsv[r];

            // ---- O += P * V : 4 d-tiles x 2 K-chunks ----
#pragma unroll
            for (int t = 0; t < 4; ++t)
#pragma unroll
                for (int c = 0; c < 2; ++c) {
                    Pack16 vb;
                    const v8bf* p = (const v8bf*)&Vt[cur][t * 16 + lo][c * 32 + hi * 16];
                    vb.h8[0] = p[0]; vb.h8[1] = p[1];
                    o[t] = __builtin_amdgcn_wmma_f32_16x16x32_bf16(
                        false, pa[c].v, false, vb.v, (short)0, o[t], false, false);
                }
        }
        __syncthreads();
    }

    // ---- normalize and store [B, L, H, D] fp32 ----
    float invl[8];
#pragma unroll
    for (int r = 0; r < 8; ++r) invl[r] = FAST_RCP(lrow[r]);
#pragma unroll
    for (int t = 0; t < 4; ++t)
#pragma unroll
        for (int r = 0; r < 8; ++r) {
            const int m = r + 8 * hi;
            const int d = t * 16 + lo;
            const size_t off = ((size_t)((size_t)b * PL + (q0 + m)) * PH + h) * PE + d;
            Out[off] = o[t][r] * invl[r];
        }
}

extern "C" void kernel_launch(void* const* d_in, const int* in_sizes, int n_in,
                              void* d_out, int out_size, void* d_ws, size_t ws_size,
                              hipStream_t stream) {
    (void)in_sizes; (void)n_in; (void)out_size; (void)d_ws; (void)ws_size;
    const float* Q = (const float*)d_in[0];
    const float* K = (const float*)d_in[1];
    const float* V = (const float*)d_in[2];
    float* O = (float*)d_out;
    const int grid = PB * PH * (PL / 128);   // 1024 workgroups, 8 waves each
    fa_causal_wmma_bf16<<<grid, 256, 0, stream>>>(Q, K, V, O);
}
